// SentAttNet_52106543235506
// MI455X (gfx1250) — matrix operand order repaired and checked by reading, not compiled, not run on hardware
//
#include <hip/hip_runtime.h>
#include <hip/hip_bf16.h>

// ---------------------------------------------------------------------------
// Types for CDNA5 WMMA (wave32, 16x16x32 bf16 -> f32)
// ---------------------------------------------------------------------------
typedef __bf16 bf16_t;
typedef __attribute__((ext_vector_type(8)))  bf16_t v8bf;
typedef __attribute__((ext_vector_type(16))) bf16_t v16bf;
typedef __attribute__((ext_vector_type(8)))  float  v8f;

#define S_LEN  64
#define BATCH  256
#define D_INP  200
#define HID    100
#define G4     400           // 4*H gate width
#define GPAD   448           // padded gate dim (weight rows / pre cols), 7*64
#define GP_LDS 416           // gate cols owned by one recurrence block (13 waves * 32)
#define KXP    224           // padded input K (200 -> 7*32)
#define KHP    128           // padded hidden K (100 -> 4*32)
#define WIN    11
#define NROW1  (S_LEN*BATCH) // 16384
#define NROW3  (WIN*NROW1)   // 180224

__device__ __forceinline__ bf16_t f2bf(float f) {
  unsigned u = __builtin_bit_cast(unsigned, f);
  u += 0x7FFFu + ((u >> 16) & 1u);             // round-to-nearest-even
  unsigned short h = (unsigned short)(u >> 16);
  return __builtin_bit_cast(bf16_t, h);
}

// A fragment (16x32, 16-bit): lane (r,half) holds row r, K = half*16..+15
__device__ __forceinline__ v16bf load_a_frag(const bf16_t* __restrict__ row_k0, int half) {
  v8bf lo = *(const v8bf*)(row_k0 + half * 16);
  v8bf hi = *(const v8bf*)(row_k0 + half * 16 + 8);
  return __builtin_shufflevector(lo, hi, 0,1,2,3,4,5,6,7,8,9,10,11,12,13,14,15);
}
// B fragment (32x16, 16-bit): lane (r,half) holds col r, K = half*16..+15 (contiguous in W[N,K])
__device__ __forceinline__ v16bf load_b_frag(const bf16_t* __restrict__ row_k0, int half) {
  return *(const v16bf*)(row_k0 + half * 16);
}
__device__ __forceinline__ v8f wmma_bf16(v16bf a, v16bf b, v8f c) {
  return __builtin_amdgcn_wmma_f32_16x16x32_bf16(false, a, false, b, (short)0, c, false, false);
}
__device__ __forceinline__ float sigm(float x) { return 1.0f / (1.0f + expf(-x)); }

// ---------------------------------------------------------------------------
// f32 -> padded bf16 (zero fill rows >= Min, cols >= Kin)
// ---------------------------------------------------------------------------
__global__ void __launch_bounds__(256)
k_pad_bf16(const float* __restrict__ in, bf16_t* __restrict__ out,
           int Min, int Kin, int Mout, int KP) {
  long total = (long)Mout * KP;
  for (long i = (long)blockIdx.x * blockDim.x + threadIdx.x; i < total;
       i += (long)gridDim.x * blockDim.x) {
    int row = (int)(i / KP), col = (int)(i % KP);
    float v = (row < Min && col < Kin) ? in[(long)row * Kin + col] : 0.0f;
    out[i] = f2bf(v);
  }
}

// ---------------------------------------------------------------------------
// C[M x GPAD] = A[M x KP](bf16) * W[GPAD x KP]^T(bf16) + bias
// block: 256 thr = 8 waves (4M x 2N), block tile 128x64, wave tile 32x32
// ---------------------------------------------------------------------------
__global__ void __launch_bounds__(256)
k_gemm_bias(const bf16_t* __restrict__ A, const bf16_t* __restrict__ W,
            const float* __restrict__ bias, float* __restrict__ C,
            int Nreal, int KP, int ldc) {
  const int tid = threadIdx.x, lane = tid & 31, wv = tid >> 5;
  const int wm = wv >> 1, wn = wv & 1;
  const int half = lane >> 4, r = lane & 15;
  const int m0 = blockIdx.x * 128 + wm * 32;
  const int n0 = blockIdx.y * 64 + wn * 32;
  v8f c00 = {}, c01 = {}, c10 = {}, c11 = {};
  const bf16_t* a0p = A + (size_t)(m0 + r) * KP;
  const bf16_t* a1p = A + (size_t)(m0 + 16 + r) * KP;
  const bf16_t* b0p = W + (size_t)(n0 + r) * KP;
  const bf16_t* b1p = W + (size_t)(n0 + 16 + r) * KP;
  for (int k0 = 0; k0 < KP; k0 += 32) {
    __builtin_prefetch(a0p + k0 + 64, 0, 1);
    v16bf a0 = load_a_frag(a0p + k0, half);
    v16bf a1 = load_a_frag(a1p + k0, half);
    v16bf b0 = load_b_frag(b0p + k0, half);
    v16bf b1 = load_b_frag(b1p + k0, half);
    c00 = wmma_bf16(a0, b0, c00);
    c01 = wmma_bf16(a0, b1, c01);
    c10 = wmma_bf16(a1, b0, c10);
    c11 = wmma_bf16(a1, b1, c11);
  }
  // D layout: lane holds col n0+j*16+r, rows half*8 + rr
#pragma unroll
  for (int i = 0; i < 2; ++i) {
    int mrow = m0 + i * 16 + half * 8;
#pragma unroll
    for (int j = 0; j < 2; ++j) {
      int col = n0 + j * 16 + r;
      float bv = (col < Nreal) ? bias[col] : 0.0f;
      v8f acc = (i == 0) ? (j == 0 ? c00 : c01) : (j == 0 ? c10 : c11);
#pragma unroll
      for (int rr = 0; rr < 8; ++rr)
        C[(size_t)(mrow + rr) * ldc + col] = acc[rr] + bv;
    }
  }
}

// ---------------------------------------------------------------------------
// LSTM recurrence. Block = 416 thr = 13 waves; owns 16 batch rows x all 416
// (padded) gate cols.  g = [x_t @ Wih^T] + h @ Whh^T + (pre | bias), then
// gate nonlinearity; h kept in LDS as bf16 (WMMA A operand), c in registers.
// ---------------------------------------------------------------------------
__global__ void __launch_bounds__(416)
k_lstm(const float* __restrict__ pre, int ldpre,          // pre-activations (or null)
       const bf16_t* __restrict__ x, int kxp,             // fused input [T,N,kxp] (or null)
       const bf16_t* __restrict__ wih,                    // [GPAD, kxp]
       const bf16_t* __restrict__ whh,                    // [GPAD, KHP]
       const float* __restrict__ bias,                    // [G4] (or null)
       float* __restrict__ hout, int ldh, int coloff,
       int T, int N, int reverse) {
  __shared__ float  gbuf[16 * GP_LDS];
  __shared__ bf16_t hbuf[16 * KHP];
  const int tid = threadIdx.x, lane = tid & 31, wv = tid >> 5;
  const int half = lane >> 4, r = lane & 15;
  const int rowbase = blockIdx.x * 16;
  const int ncol0 = wv * 32;
  for (int i = tid; i < 16 * KHP; i += 416) hbuf[i] = f2bf(0.0f);
  __syncthreads();
  float cst[4] = {0.f, 0.f, 0.f, 0.f};
  const bf16_t* whh0 = whh + (size_t)(ncol0 + r) * KHP;
  const bf16_t* whh1 = whh + (size_t)(ncol0 + 16 + r) * KHP;
  const bf16_t* wih0 = wih ? wih + (size_t)(ncol0 + r) * kxp : nullptr;
  const bf16_t* wih1 = wih ? wih + (size_t)(ncol0 + 16 + r) * kxp : nullptr;
  const bf16_t* hrow = &hbuf[r * KHP];

  for (int s = 0; s < T; ++s) {
    const int t = reverse ? (T - 1 - s) : s;
    v8f acc0 = {}, acc1 = {};
#pragma unroll
    for (int k0 = 0; k0 < KHP; k0 += 32) {                // h @ Whh^T
      v16bf a  = load_a_frag(hrow + k0, half);
      v16bf b0 = load_b_frag(whh0 + k0, half);
      v16bf b1 = load_b_frag(whh1 + k0, half);
      acc0 = wmma_bf16(a, b0, acc0);
      acc1 = wmma_bf16(a, b1, acc1);
    }
    if (x != nullptr) {                                   // fused x_t @ Wih^T
      const bf16_t* xrow = x + ((size_t)t * N + rowbase + r) * kxp;
      for (int k0 = 0; k0 < kxp; k0 += 32) {
        v16bf a  = load_a_frag(xrow + k0, half);
        v16bf b0 = load_b_frag(wih0 + k0, half);
        v16bf b1 = load_b_frag(wih1 + k0, half);
        acc0 = wmma_bf16(a, b0, acc0);
        acc1 = wmma_bf16(a, b1, acc1);
      }
    }
#pragma unroll
    for (int j = 0; j < 2; ++j) {                         // spill gate tile to LDS
      int col = ncol0 + j * 16 + r;
      v8f acc = j ? acc1 : acc0;
      float bv = (bias != nullptr && col < G4) ? bias[col] : 0.0f;
#pragma unroll
      for (int rr = 0; rr < 8; ++rr) {
        int row = half * 8 + rr;
        float v = acc[rr] + bv;
        if (pre != nullptr)
          v += pre[((size_t)t * N + rowbase + row) * ldpre + col];
        gbuf[row * GP_LDS + col] = v;
      }
    }
    __syncthreads();
#pragma unroll
    for (int k = 0; k < 4; ++k) {                         // gate nonlinearity
      int e = tid + k * 416;
      if (e < 16 * HID) {
        int row = e / HID, j = e - row * HID;
        float ig = gbuf[row * GP_LDS + j];
        float fg = gbuf[row * GP_LDS + HID + j];
        float gg = gbuf[row * GP_LDS + 2 * HID + j];
        float og = gbuf[row * GP_LDS + 3 * HID + j];
        float cc = sigm(fg) * cst[k] + sigm(ig) * tanhf(gg);
        cst[k] = cc;
        float hv = sigm(og) * tanhf(cc);
        hbuf[row * KHP + j] = f2bf(hv);
        hout[((size_t)t * N + rowbase + row) * ldh + coloff + j] = hv;
      }
    }
    __syncthreads();
  }
}

// ---------------------------------------------------------------------------
// Window attention scores + softmax: one wave per token n = b*S + s.
// att[n,w] = softmax_w( q[n] . src[s+w-5, b] ), zero score for OOB slots.
// ---------------------------------------------------------------------------
__global__ void __launch_bounds__(256)
k_attn(const float* __restrict__ q, const float* __restrict__ src,
       float* __restrict__ att) {
  const int lane = threadIdx.x & 31, wv = threadIdx.x >> 5;
  const int n = blockIdx.x * 8 + wv;
  const int s = n & 63, b = n >> 6;
  float sc[WIN];
#pragma unroll
  for (int w = 0; w < WIN; ++w) sc[w] = 0.0f;
  const float* qr = q + ((size_t)s * BATCH + b) * 200;
  for (int d = lane; d < 200; d += 32) {
    float qv = qr[d];
#pragma unroll
    for (int w = 0; w < WIN; ++w) {
      int ts = s + w - 5;
      if (ts >= 0 && ts < S_LEN)
        sc[w] += qv * src[((size_t)ts * BATCH + b) * 200 + d];
    }
  }
#pragma unroll
  for (int off = 16; off > 0; off >>= 1)
#pragma unroll
    for (int w = 0; w < WIN; ++w) sc[w] += __shfl_xor(sc[w], off, 32);
  float m = sc[0];
#pragma unroll
  for (int w = 1; w < WIN; ++w) m = fmaxf(m, sc[w]);
  float sum = 0.0f;
#pragma unroll
  for (int w = 0; w < WIN; ++w) { sc[w] = expf(sc[w] - m); sum += sc[w]; }
  float inv = 1.0f / sum;
  if (lane == 0) {
#pragma unroll
    for (int w = 0; w < WIN; ++w) att[(size_t)n * WIN + w] = sc[w] * inv;
  }
}

// ---------------------------------------------------------------------------
// Build attention-weighted window rows as padded bf16 [WIN, NROW1, KXP]
// (directly in WMMA-A-ready layout for the fused lstm3/4 kernels).
// ---------------------------------------------------------------------------
__global__ void __launch_bounds__(256)
k_build_fo(const float* __restrict__ src, const float* __restrict__ att,
           bf16_t* __restrict__ fo) {
  const int lane = threadIdx.x & 31, wv = threadIdx.x >> 5;
  const long row = (long)blockIdx.x * 8 + wv;   // row in [0, WIN*NROW1)
  const int w = (int)(row / NROW1);
  const int n = (int)(row - (long)w * NROW1);
  const int s = n & 63, b = n >> 6, ts = s + w - 5;
  const float a = att[(size_t)n * WIN + w];
  bf16_t* dst = fo + row * KXP;
  const bool ok = (ts >= 0 && ts < S_LEN);
  const float* sr = src + ((size_t)(ok ? ts : 0) * BATCH + b) * 200;
  for (int kp = lane; kp < KXP; kp += 32) {
    float v = (ok && kp < 200) ? sr[kp] * a : 0.0f;
    dst[kp] = f2bf(v);
  }
}

// ---------------------------------------------------------------------------
// 2-wide FC head (+ optional transpose to [B,S,2] and sel = out1>out0)
// ---------------------------------------------------------------------------
__global__ void __launch_bounds__(256)
k_fc2(const float* __restrict__ h, const float* __restrict__ w2,
      const float* __restrict__ b2, float* __restrict__ out,
      float* __restrict__ sel, int M, int transpose) {
  const int lane = threadIdx.x & 31, wv = threadIdx.x >> 5;
  const int row = blockIdx.x * 8 + wv;
  if (row >= M) return;
  const float* hr = h + (size_t)row * 200;
  float a0 = 0.f, a1 = 0.f;
  for (int d = lane; d < 200; d += 32) {
    float hv = hr[d];
    a0 += hv * w2[d];
    a1 += hv * w2[200 + d];
  }
#pragma unroll
  for (int off = 16; off > 0; off >>= 1) {
    a0 += __shfl_xor(a0, off, 32);
    a1 += __shfl_xor(a1, off, 32);
  }
  if (lane == 0) {
    a0 += b2[0]; a1 += b2[1];
    size_t o;
    if (transpose) {
      int ss = row >> 8, bb = row & 255;              // row = s*B + b
      size_t ridx = (size_t)bb * S_LEN + ss;
      o = ridx * 2;
      if (sel) sel[ridx] = (a1 > a0) ? 1.0f : 0.0f;
    } else {
      o = (size_t)row * 2;
    }
    out[o] = a0; out[o + 1] = a1;
  }
}

// ---------------------------------------------------------------------------
// Host orchestration
// ---------------------------------------------------------------------------
extern "C" void kernel_launch(void* const* d_in, const int* in_sizes, int n_in,
                              void* d_out, int out_size, void* d_ws, size_t ws_size,
                              hipStream_t stream) {
  (void)in_sizes; (void)n_in; (void)out_size; (void)ws_size;
  const float* input1 = (const float*)d_in[0];
  const float* input2 = (const float*)d_in[1];
  // params start at d_in[6]: per lstm{1..4} x dir{f,b}: wih, whh, b
  auto P = [&](int lstm, int dir, int which) -> const float* {
    return (const float*)d_in[6 + lstm * 6 + dir * 3 + which];
  };
  const float *fcw[4], *fcb[4];
  for (int i = 0; i < 4; ++i) {
    fcw[i] = (const float*)d_in[30 + 2 * i];
    fcb[i] = (const float*)d_in[31 + 2 * i];
  }

  char* ws = (char*)d_ws;
  size_t off = 0;
  auto alloc = [&](size_t bytes) -> void* {
    void* p = ws + off;
    off += (bytes + 255) & ~(size_t)255;
    return p;
  };
  bf16_t* x1b = (bf16_t*)alloc((size_t)NROW1 * KXP * 2);
  bf16_t* x2b = (bf16_t*)alloc((size_t)NROW1 * KXP * 2);
  bf16_t *wihb[8], *whhb[8];
  for (int i = 0; i < 8; ++i) {
    wihb[i] = (bf16_t*)alloc((size_t)GPAD * KXP * 2);
    whhb[i] = (bf16_t*)alloc((size_t)GPAD * KHP * 2);
  }
  float* pre[4];
  for (int i = 0; i < 4; ++i) pre[i] = (float*)alloc((size_t)NROW1 * GPAD * 4);
  float* f1   = (float*)alloc((size_t)NROW1 * 200 * 4);
  float* f3   = (float*)alloc((size_t)NROW1 * 200 * 4);
  float* att1 = (float*)alloc((size_t)NROW1 * WIN * 4);
  float* att2 = (float*)alloc((size_t)NROW1 * WIN * 4);
  bf16_t* fo1 = (bf16_t*)alloc((size_t)NROW3 * KXP * 2);
  bf16_t* fo2 = (bf16_t*)alloc((size_t)NROW3 * KXP * 2);
  float* h3   = (float*)alloc((size_t)NROW3 * 200 * 4);
  float* h4   = (float*)alloc((size_t)NROW3 * 200 * 4);

  // 1) convert inputs + weights to padded bf16
  k_pad_bf16<<<4096, 256, 0, stream>>>(input1, x1b, NROW1, D_INP, NROW1, KXP);
  k_pad_bf16<<<4096, 256, 0, stream>>>(input2, x2b, NROW1, D_INP, NROW1, KXP);
  for (int i = 0; i < 8; ++i) {
    int lstm = i >> 1, dir = i & 1;
    k_pad_bf16<<<392, 256, 0, stream>>>(P(lstm, dir, 0), wihb[i], G4, D_INP, GPAD, KXP);
    k_pad_bf16<<<224, 256, 0, stream>>>(P(lstm, dir, 1), whhb[i], G4, HID,   GPAD, KHP);
  }

  // 2) big input-projection GEMMs for lstm1/2 (bias folded in)
  dim3 gg(NROW1 / 128, GPAD / 64);
  k_gemm_bias<<<gg, 256, 0, stream>>>(x1b, wihb[0], P(0,0,2), pre[0], G4, KXP, GPAD);
  k_gemm_bias<<<gg, 256, 0, stream>>>(x1b, wihb[1], P(0,1,2), pre[1], G4, KXP, GPAD);
  k_gemm_bias<<<gg, 256, 0, stream>>>(x2b, wihb[2], P(1,0,2), pre[2], G4, KXP, GPAD);
  k_gemm_bias<<<gg, 256, 0, stream>>>(x2b, wihb[3], P(1,1,2), pre[3], G4, KXP, GPAD);

  // 3) lstm1/2 recurrences -> f1, f3 (forward cols 0..99, backward 100..199)
  k_lstm<<<BATCH/16, 416, 0, stream>>>(pre[0], GPAD, nullptr, 0, nullptr, whhb[0], nullptr, f1, 200, 0,   S_LEN, BATCH, 0);
  k_lstm<<<BATCH/16, 416, 0, stream>>>(pre[1], GPAD, nullptr, 0, nullptr, whhb[1], nullptr, f1, 200, 100, S_LEN, BATCH, 1);
  k_lstm<<<BATCH/16, 416, 0, stream>>>(pre[2], GPAD, nullptr, 0, nullptr, whhb[2], nullptr, f3, 200, 0,   S_LEN, BATCH, 0);
  k_lstm<<<BATCH/16, 416, 0, stream>>>(pre[3], GPAD, nullptr, 0, nullptr, whhb[3], nullptr, f3, 200, 100, S_LEN, BATCH, 1);

  float* out  = (float*)d_out;
  float* out1 = out;                 // [B,S,2]   32768
  float* out2 = out + 32768;         // [B,S,2]   32768
  float* out3 = out + 65536;         // [W,BS,2] 360448
  float* sel1 = out + 425984;        // [BS]      16384
  float* out4 = out + 442368;        // [W,BS,2] 360448
  float* sel2 = out + 802816;        // [BS]      16384

  // 4) out1/out2 + selection masks
  k_fc2<<<NROW1/8, 256, 0, stream>>>(f1, fcw[0], fcb[0], out1, sel1, NROW1, 1);
  k_fc2<<<NROW1/8, 256, 0, stream>>>(f3, fcw[1], fcb[1], out2, sel2, NROW1, 1);

  // 5) windowed attention + weighted bf16 windows
  k_attn<<<NROW1/8, 256, 0, stream>>>(f1, f3, att1);
  k_attn<<<NROW1/8, 256, 0, stream>>>(f3, f1, att2);
  k_build_fo<<<NROW3/8, 256, 0, stream>>>(f3, att1, fo1);
  k_build_fo<<<NROW3/8, 256, 0, stream>>>(f1, att2, fo2);

  // 6) lstm3/4 with fused per-step input projection (batch 16384 per step)
  k_lstm<<<NROW1/16, 416, 0, stream>>>(nullptr, 0, fo1, KXP, wihb[4], whhb[4], P(2,0,2), h3, 200, 0,   WIN, NROW1, 0);
  k_lstm<<<NROW1/16, 416, 0, stream>>>(nullptr, 0, fo1, KXP, wihb[5], whhb[5], P(2,1,2), h3, 200, 100, WIN, NROW1, 1);
  k_lstm<<<NROW1/16, 416, 0, stream>>>(nullptr, 0, fo2, KXP, wihb[6], whhb[6], P(3,0,2), h4, 200, 0,   WIN, NROW1, 0);
  k_lstm<<<NROW1/16, 416, 0, stream>>>(nullptr, 0, fo2, KXP, wihb[7], whhb[7], P(3,1,2), h4, 200, 100, WIN, NROW1, 1);

  // 7) out3/out4
  k_fc2<<<NROW3/8, 256, 0, stream>>>(h3, fcw[2], fcb[2], out3, nullptr, NROW3, 0);
  k_fc2<<<NROW3/8, 256, 0, stream>>>(h4, fcw[3], fcb[3], out4, nullptr, NROW3, 0);
}
